// GNS_38878043963418
// MI455X (gfx1250) — matrix operand order repaired and checked by reading, not compiled, not run on hardware
//
#include <hip/hip_runtime.h>

// ---------------------------------------------------------------------------
// GNS message-passing network for MI455X (gfx1250), wave32 + WMMA.
//
// Roofline: ~270 GFLOP of 64-wide GEMMs vs ~3.3 GB HBM traffic (edge state
// read+write per block; node/agg tables are L2-resident: 12.8MB each vs 192MB
// L2). At 23.3 TB/s the memory floor is ~140us, so the GEMMs must ride
// v_wmma_f32_16x16x32_f16 (f16 in, f32 accumulate; numerically safe because
// every MLP output is LayerNormed). proc_edge is never materialized: the
// edge-MLP epilogue fuses the residual add and the segment_sum
// (global_atomic_add_f32 into the L2-resident agg table).
//
// Tiling: 32 rows x 64 cols per workgroup, 4 wave32s, one 16-col N-tile per
// wave; each wave holds one B fragment in registers and issues two WMMAs
// (two 16-row M-subtiles) -> 2x B reuse. All LDS tiles use padded strides so
// the per-lane ds_load_b128 A-fragment reads are 64-bank conflict-free
// (stride mod 64 DW generates 16 distinct 4-bank groups).
// ---------------------------------------------------------------------------

typedef __attribute__((ext_vector_type(16))) _Float16 v16h;
typedef __attribute__((ext_vector_type(8)))  _Float16 v8h;
typedef __attribute__((ext_vector_type(8)))  float    v8f;

// A fragment (16x32 f16, MxK) from a row-major f16 tile with stride_h halves.
// Lane = (half, m): elements 0..15 hold K = k0 + half*16 + e.
// Both halves are 8 contiguous f16 -> two ds_load_b128.
__device__ __forceinline__ v16h frag_a(const _Float16* base, int stride_h,
                                       int m, int k0, int half) {
  const v8h* p0 = (const v8h*)(base + m * stride_h + k0 + half * 16);
  const v8h* p1 = (const v8h*)(base + m * stride_h + k0 + half * 16 + 8);
  v8h a0 = *p0, a1 = *p1;
  v16h a;
#pragma unroll
  for (int i = 0; i < 8; ++i) { a[i] = a0[i]; a[i + 8] = a1[i]; }
  return a;
}

// B fragment (32x16 f16, KxN) from weights pre-packed in fragment order:
// linear index ((nt*kc + c)*32 + lane)*16 + e, where lane=(half,n) holds
// K = c*32 + half*16 + e for column n. 32 contiguous bytes per lane.
__device__ __forceinline__ v16h frag_b(const _Float16* __restrict__ wp,
                                       int kc, int c, int nt, int lane) {
  const v8h* p = (const v8h*)(wp + (((size_t)(nt * kc + c) * 32 + lane) << 4));
  v8h b0 = p[0], b1 = p[1];
  v16h b;
#pragma unroll
  for (int i = 0; i < 8; ++i) { b[i] = b0[i]; b[i + 8] = b1[i]; }
  return b;
}

// Pack a (din x 64) f32 weight matrix into f16 B-fragment order, zero-padding
// K to kc*32. out[tid] with tid = ((nt*kc + c)*32 + l)*16 + e.
__global__ void pack_b_kernel(const float* __restrict__ W,
                              _Float16* __restrict__ out, int din, int kc) {
  int tid = blockIdx.x * 256 + threadIdx.x;
  int total = kc * 2048;
  if (tid >= total) return;
  int e  = tid & 15;
  int l  = (tid >> 4) & 31;
  int c  = (tid >> 9) % kc;
  int nt = tid / (512 * kc);
  int k  = c * 32 + (l >> 4) * 16 + e;
  int n  = nt * 16 + (l & 15);
  float v = (k < din) ? W[(size_t)k * 64 + n] : 0.0f;
  out[tid] = (_Float16)v;
}

__global__ void zero_kernel(float* __restrict__ p, int n) {
  int i = blockIdx.x * 256 + threadIdx.x;
  if (i < n) p[i] = 0.0f;
}

// ---------------------------------------------------------------------------
// Fused 3-layer MLP + LayerNorm over a 32-row tile.
// IN_MODE:  0 = plain DIN-column input matrix (encoders)
//           1 = concat(edge[e], node[recv[e]], node[send[e]])  (din=192)
//           2 = concat(node[r], agg[r], global_feat[r])        (din=130)
// OUT_MODE: 0 = store y                                  (encoders)
//           1 = edge += y ; atomicAdd(agg[recv], y)      (edge block)
//           2 = node += y                                (node block)
// ---------------------------------------------------------------------------
template <int DIN, int KPAD, int IN_MODE, int OUT_MODE>
__global__ __launch_bounds__(128) void mlp_ln_kernel(
    const float* __restrict__ in0, const float* __restrict__ in1,
    const float* __restrict__ in2, const int* __restrict__ eidx,
    float* __restrict__ outbuf, float* __restrict__ aggbuf,
    const _Float16* __restrict__ w1p, const _Float16* __restrict__ w2p,
    const _Float16* __restrict__ w3p,
    const float* __restrict__ b1, const float* __restrict__ b2,
    const float* __restrict__ b3,
    const float* __restrict__ gam, const float* __restrict__ bet, int rows) {
  static_assert(KPAD % 32 == 0, "KPAD must be a multiple of 32");
  constexpr int KC1 = KPAD / 32;
  constexpr int AST = KPAD + 8;  // f16 stride: (AST/2) mod 64 DW -> conflict-free
  constexpr int HST = 72;        // f16 stride for H1/H2 (36 DW pattern)
  constexpr int QST = 65;        // f32 stride for H3 (m*65 mod 64 = m)

  __shared__ __align__(16) _Float16 As[32 * AST];
  __shared__ __align__(16) _Float16 H1[32 * HST];
  __shared__ __align__(16) _Float16 H2[32 * HST];
  __shared__ float H3[32 * QST];
  __shared__ float redS[32 * 4];
  __shared__ float redQ[32 * 4];
  __shared__ float muS[32], rsS[32];
  __shared__ int recvL[32], sendL[32];

  const int tid  = threadIdx.x;
  const int lane = tid & 31;
  const int nt   = tid >> 5;   // wave id == N-tile
  const int half = lane >> 4;
  const int l15  = lane & 15;  // m within subtile for A, n for B/D
  const int col  = nt * 16 + l15;
  const int row0 = blockIdx.x * 32;

  if (IN_MODE == 1) {
    if (tid < 32) {
      int e = row0 + tid;
      int re = 0, se = 0;
      if (e < rows) { re = eidx[2 * e]; se = eidx[2 * e + 1]; }
      recvL[tid] = re; sendL[tid] = se;
    }
    __syncthreads();
  }

  // ---- stage A tile (f32 -> f16) into LDS, zero-padded to KPAD columns ----
  for (int idx = tid; idx < 32 * KPAD; idx += 128) {
    int r = idx / KPAD, c = idx % KPAD;
    int row = row0 + r;
    float v = 0.0f;
    if (row < rows) {
      if (IN_MODE == 0) {
        if (c < DIN) v = in0[(size_t)row * DIN + c];
      } else if (IN_MODE == 1) {
        if (c < 64)       v = in0[(size_t)row * 64 + c];
        else if (c < 128) v = in1[(size_t)recvL[r] * 64 + (c - 64)];
        else              v = in1[(size_t)sendL[r] * 64 + (c - 128)];
      } else {
        if (c < 64)       v = in0[(size_t)row * 64 + c];
        else if (c < 128) v = in1[(size_t)row * 64 + (c - 64)];
        else if (c < 130) v = in2[(size_t)row * 2 + (c - 128)];
      }
    }
    As[r * AST + c] = (_Float16)v;
  }
  __syncthreads();

  // ---- layer 1: (32 x KPAD) @ (KPAD x 64) + b1, relu -> H1 (f16) ----
  {
    v8f acc0 = {}, acc1 = {};
#pragma unroll
    for (int c = 0; c < KC1; ++c) {
      v16h b  = frag_b(w1p, KC1, c, nt, lane);
      v16h a0 = frag_a(As, AST, l15, c * 32, half);
      v16h a1 = frag_a(As + 16 * AST, AST, l15, c * 32, half);
      acc0 = __builtin_amdgcn_wmma_f32_16x16x32_f16(false, a0, false, b,
                                                    (short)0, acc0, false, false);
      acc1 = __builtin_amdgcn_wmma_f32_16x16x32_f16(false, a1, false, b,
                                                    (short)0, acc1, false, false);
    }
    float bb = b1[col];
#pragma unroll
    for (int g = 0; g < 8; ++g) {
      float v0 = acc0[g] + bb, v1 = acc1[g] + bb;
      v0 = v0 > 0.0f ? v0 : 0.0f;
      v1 = v1 > 0.0f ? v1 : 0.0f;
      H1[(g + 8 * half) * HST + col]        = (_Float16)v0;
      H1[(16 + g + 8 * half) * HST + col]   = (_Float16)v1;
    }
  }
  __syncthreads();

  // ---- layer 2: (32 x 64) @ (64 x 64) + b2, relu -> H2 (f16) ----
  {
    v8f acc0 = {}, acc1 = {};
#pragma unroll
    for (int c = 0; c < 2; ++c) {
      v16h b  = frag_b(w2p, 2, c, nt, lane);
      v16h a0 = frag_a(H1, HST, l15, c * 32, half);
      v16h a1 = frag_a(H1 + 16 * HST, HST, l15, c * 32, half);
      acc0 = __builtin_amdgcn_wmma_f32_16x16x32_f16(false, a0, false, b,
                                                    (short)0, acc0, false, false);
      acc1 = __builtin_amdgcn_wmma_f32_16x16x32_f16(false, a1, false, b,
                                                    (short)0, acc1, false, false);
    }
    float bb = b2[col];
#pragma unroll
    for (int g = 0; g < 8; ++g) {
      float v0 = acc0[g] + bb, v1 = acc1[g] + bb;
      v0 = v0 > 0.0f ? v0 : 0.0f;
      v1 = v1 > 0.0f ? v1 : 0.0f;
      H2[(g + 8 * half) * HST + col]        = (_Float16)v0;
      H2[(16 + g + 8 * half) * HST + col]   = (_Float16)v1;
    }
  }
  __syncthreads();

  // ---- layer 3: (32 x 64) @ (64 x 64) + b3 -> H3 (f32, pre-LN) ----
  {
    v8f acc0 = {}, acc1 = {};
#pragma unroll
    for (int c = 0; c < 2; ++c) {
      v16h b  = frag_b(w3p, 2, c, nt, lane);
      v16h a0 = frag_a(H2, HST, l15, c * 32, half);
      v16h a1 = frag_a(H2 + 16 * HST, HST, l15, c * 32, half);
      acc0 = __builtin_amdgcn_wmma_f32_16x16x32_f16(false, a0, false, b,
                                                    (short)0, acc0, false, false);
      acc1 = __builtin_amdgcn_wmma_f32_16x16x32_f16(false, a1, false, b,
                                                    (short)0, acc1, false, false);
    }
    float bb = b3[col];
#pragma unroll
    for (int g = 0; g < 8; ++g) {
      H3[(g + 8 * half) * QST + col]      = acc0[g] + bb;
      H3[(16 + g + 8 * half) * QST + col] = acc1[g] + bb;
    }
  }
  __syncthreads();

  // ---- LayerNorm over 64 columns per row: 4 threads x 16 elems per row ----
  {
    int r = tid >> 2, q = tid & 3;
    float s = 0.0f, ss = 0.0f;
#pragma unroll
    for (int jj = 0; jj < 16; ++jj) {
      float v = H3[r * QST + q * 16 + jj];
      s += v; ss += v * v;
    }
    redS[r * 4 + q] = s;
    redQ[r * 4 + q] = ss;
  }
  __syncthreads();
  if (tid < 32) {
    float s = 0.0f, ss = 0.0f;
#pragma unroll
    for (int q = 0; q < 4; ++q) { s += redS[tid * 4 + q]; ss += redQ[tid * 4 + q]; }
    float mu  = s * (1.0f / 64.0f);
    float var = ss * (1.0f / 64.0f) - mu * mu;
    muS[tid] = mu;
    rsS[tid] = rsqrtf(var + 1e-5f);
  }
  __syncthreads();

  // ---- epilogue: y = (h - mu)*rsig*g + be, then mode-specific writeback ----
  {
    int r = tid >> 2, q = tid & 3;
    int row = row0 + r;
    if (row < rows) {
      float mu = muS[r], rs = rsS[r];
#pragma unroll
      for (int jj = 0; jj < 16; ++jj) {
        int j = q * 16 + jj;
        float y = (H3[r * QST + j] - mu) * rs * gam[j] + bet[j];
        if (OUT_MODE == 0) {
          outbuf[(size_t)row * 64 + j] = y;
        } else if (OUT_MODE == 1) {
          outbuf[(size_t)row * 64 + j] += y;                  // edge += proc_edge
          atomicAdd(&aggbuf[(size_t)recvL[r] * 64 + j], y);   // segment_sum
        } else {
          outbuf[(size_t)row * 64 + j] += y;                  // node += proc_node
        }
      }
    }
  }
}

// ---------------------------------------------------------------------------
// Decoder: 64 -> 64 (relu) -> 32 (relu) -> 3. ~0.3 GFLOP total: plain LDS.
// ---------------------------------------------------------------------------
__global__ __launch_bounds__(64) void dec_kernel(
    const float* __restrict__ node,
    const float* __restrict__ w1, const float* __restrict__ b1,
    const float* __restrict__ w2, const float* __restrict__ b2,
    const float* __restrict__ w3, const float* __restrict__ b3,
    float* __restrict__ out, int N) {
  __shared__ float xs[64], h1[64], h2[32];
  int row = blockIdx.x;
  int j = threadIdx.x;
  xs[j] = node[(size_t)row * 64 + j];
  __syncthreads();
  float a = b1[j];
  for (int k = 0; k < 64; ++k) a += xs[k] * w1[k * 64 + j];
  h1[j] = a > 0.0f ? a : 0.0f;
  __syncthreads();
  if (j < 32) {
    float a2 = b2[j];
    for (int k = 0; k < 64; ++k) a2 += h1[k] * w2[k * 32 + j];
    h2[j] = a2 > 0.0f ? a2 : 0.0f;
  }
  __syncthreads();
  if (j < 3) {
    float a3 = b3[j];
    for (int k = 0; k < 32; ++k) a3 += h2[k] * w3[k * 3 + j];
    out[(size_t)row * 3 + j] = a3;
  }
}

// ---------------------------------------------------------------------------
// Host side. Param leaves in setup_inputs() insertion order:
//  0 node_feat  1 edge_feat  2 global_feat  3 edge_idx  4 node_size
//  5..12  node_enc {w1,b1,w2,b2,w3,b3,g,be}
// 13..20  edge_enc {...}
// 21+16i+{0..7}  block i edge MLP; 21+16i+{8..15} block i node MLP
// 149..154 dec {w1,b1,w2,b2,w3,b3}
// ---------------------------------------------------------------------------
extern "C" void kernel_launch(void* const* d_in, const int* in_sizes, int n_in,
                              void* d_out, int out_size, void* d_ws,
                              size_t ws_size, hipStream_t stream) {
  (void)n_in; (void)out_size; (void)ws_size;
  const int N = in_sizes[0] / 17;
  const int E = in_sizes[1] / 4;

  const float* node_feat   = (const float*)d_in[0];
  const float* edge_feat   = (const float*)d_in[1];
  const float* global_feat = (const float*)d_in[2];
  const int*   eidx        = (const int*)d_in[3];
  auto F = [&](int i) { return (const float*)d_in[i]; };

  char* ws = (char*)d_ws;
  size_t off = 0;
  auto alloc = [&](size_t bytes) {
    size_t o = off;
    off = (off + bytes + 255) & ~(size_t)255;
    return o;
  };
  float* nodebuf = (float*)(ws + alloc((size_t)N * 64 * 4));
  float* edgebuf = (float*)(ws + alloc((size_t)E * 64 * 4));
  float* aggbuf  = (float*)(ws + alloc((size_t)N * 64 * 4));

  auto packW = [&](int src_idx, int din, int kc) -> _Float16* {
    _Float16* dst = (_Float16*)(ws + alloc((size_t)kc * 2048 * sizeof(_Float16)));
    int total = kc * 2048;
    pack_b_kernel<<<(total + 255) / 256, 256, 0, stream>>>(
        (const float*)d_in[src_idx], dst, din, kc);
    return dst;
  };

  _Float16 *ne_w1 = packW(5, 17, 1),  *ne_w2 = packW(7, 64, 2),  *ne_w3 = packW(9, 64, 2);
  _Float16 *ee_w1 = packW(13, 4, 1),  *ee_w2 = packW(15, 64, 2), *ee_w3 = packW(17, 64, 2);
  _Float16 *be_w1[8], *be_w2[8], *be_w3[8], *bn_w1[8], *bn_w2[8], *bn_w3[8];
  for (int i = 0; i < 8; ++i) {
    int eb = 21 + 16 * i, nb = eb + 8;
    be_w1[i] = packW(eb + 0, 192, 6);
    be_w2[i] = packW(eb + 2, 64, 2);
    be_w3[i] = packW(eb + 4, 64, 2);
    bn_w1[i] = packW(nb + 0, 130, 5);
    bn_w2[i] = packW(nb + 2, 64, 2);
    bn_w3[i] = packW(nb + 4, 64, 2);
  }

  const int tiles_n = (N + 31) / 32;
  const int tiles_e = (E + 31) / 32;

  // Encoders.
  mlp_ln_kernel<17, 32, 0, 0><<<tiles_n, 128, 0, stream>>>(
      node_feat, nullptr, nullptr, nullptr, nodebuf, nullptr,
      ne_w1, ne_w2, ne_w3, F(6), F(8), F(10), F(11), F(12), N);
  mlp_ln_kernel<4, 32, 0, 0><<<tiles_e, 128, 0, stream>>>(
      edge_feat, nullptr, nullptr, nullptr, edgebuf, nullptr,
      ee_w1, ee_w2, ee_w3, F(14), F(16), F(18), F(19), F(20), E);

  // 8 message-passing blocks.
  for (int i = 0; i < 8; ++i) {
    int eb = 21 + 16 * i, nb = eb + 8;
    zero_kernel<<<(N * 64 + 255) / 256, 256, 0, stream>>>(aggbuf, N * 64);
    mlp_ln_kernel<192, 192, 1, 1><<<tiles_e, 128, 0, stream>>>(
        edgebuf, nodebuf, nullptr, eidx, edgebuf, aggbuf,
        be_w1[i], be_w2[i], be_w3[i],
        F(eb + 1), F(eb + 3), F(eb + 5), F(eb + 6), F(eb + 7), E);
    mlp_ln_kernel<130, 160, 2, 2><<<tiles_n, 128, 0, stream>>>(
        nodebuf, aggbuf, global_feat, nullptr, nodebuf, nullptr,
        bn_w1[i], bn_w2[i], bn_w3[i],
        F(nb + 1), F(nb + 3), F(nb + 5), F(nb + 6), F(nb + 7), N);
  }

  // Decoder.
  dec_kernel<<<N, 64, 0, stream>>>(nodebuf, F(149), F(150), F(151), F(152),
                                   F(153), F(154), (float*)d_out, N);
}
